// LlamaAttention_84670985273407
// MI455X (gfx1250) — compile-verified, matrix operand order checked by reading, not run on hardware
//
#include <hip/hip_runtime.h>
#include <hip/hip_bf16.h>

// ---------------------------------------------------------------------------
// Types for CDNA5 WMMA (wave32): v_wmma_f32_16x16x32_bf16
// ---------------------------------------------------------------------------
typedef __bf16 v16bf __attribute__((ext_vector_type(16)));
typedef __bf16 v8bf  __attribute__((ext_vector_type(8)));
typedef __bf16 v4bf  __attribute__((ext_vector_type(4)));
typedef float  v8f   __attribute__((ext_vector_type(8)));

static constexpr int QLEN = 2048;
static constexpr int HID  = 4096;
static constexpr int NH   = 32;
static constexpr int DH   = 128;

// ---------------------------------------------------------------------------
// CDNA5 async global->LDS copy (16B per lane), tracked with ASYNCcnt.
// VDST VGPR holds the LDS byte offset, added to the LDS base and instruction offset,
// VADDR holds the 64-bit global address.  (CDNA5 ISA)
// ---------------------------------------------------------------------------
__device__ __forceinline__ void async_ld_b128(void* lds, const void* gptr) {
    unsigned loff = (unsigned)(size_t)lds;             // low 32 bits = LDS offset
    unsigned long long ga = (unsigned long long)(size_t)gptr;
    asm volatile("global_load_async_to_lds_b128 %0, %1, off"
                 :: "v"(loff), "v"(ga) : "memory");
}
__device__ __forceinline__ void wait_async0() {
    asm volatile("s_wait_asynccnt 0x0" ::: "memory");
}

// Load a 16-bit A/B WMMA operand fragment for one lane from a row-major row.
// Per CDNA5 ISA: lane (m = lane&15, hi = lane>>4) holds halfs j=0..15 with
// k = j + (j>=8 ? 8 : 0) + hi*8  ->  two contiguous 8-element chunks at
// k = hi*8 and k = 16 + hi*8.
__device__ __forceinline__ v16bf load_frag_row(const __bf16* rowbase, int hi) {
    union { v16bf v; v8bf h[2]; } u;
    u.h[0] = *(const v8bf*)(rowbase + hi * 8);
    u.h[1] = *(const v8bf*)(rowbase + 16 + hi * 8);
    return u.v;
}

// ---------------------------------------------------------------------------
// fp32 -> bf16 conversion (hidden states), 8 elements per thread
// ---------------------------------------------------------------------------
__global__ __launch_bounds__(256) void cvt_f32_to_bf16(const float* __restrict__ in,
                                                       __bf16* __restrict__ out) {
    size_t i = ((size_t)blockIdx.x * 256 + threadIdx.x) * 8;
    float4 a = *(const float4*)&in[i];
    float4 b = *(const float4*)&in[i + 4];
    v8bf o;
    o[0] = (__bf16)a.x; o[1] = (__bf16)a.y; o[2] = (__bf16)a.z; o[3] = (__bf16)a.w;
    o[4] = (__bf16)b.x; o[5] = (__bf16)b.y; o[6] = (__bf16)b.z; o[7] = (__bf16)b.w;
    *(v8bf*)&out[i] = o;
}

// ---------------------------------------------------------------------------
// GEMM: C[M,N] = A_bf16[M,K] * W_f32[N,K]^T, M=2048, N=K=4096 (compile-time).
// BM=256, BN=128, BK=32; 256 threads = 8 waves in a 4(M) x 2(N) grid; each
// wave owns a 64x64 tile = 4x4 WMMA tiles -> 16 v_wmma per 16 ds_load_b128
// per K-step, with 4x fragment reuse.  Double-buffered LDS; A tile staged
// with async-to-LDS, W tile staged through registers with fused f32->bf16
// convert and packed b64 LDS stores.
// ---------------------------------------------------------------------------
static constexpr int BM = 256, BN = 128, BK = 32;
static constexpr int LDT = BK + 16;   // 48 bf16 = 96 B row stride (16B multiple)
static constexpr int GM = QLEN, GN = HID, GK = HID;

template <bool STORE_F32>
__global__ __launch_bounds__(256)
void gemm_bf16(const __bf16* __restrict__ A,   // M x K  (bf16)
               const float*  __restrict__ W,   // N x K  (f32)
               void* __restrict__ outp) {
    __shared__ __bf16 As[2][BM][LDT];
    __shared__ __bf16 Bs[2][BN][LDT];

    const int tid   = threadIdx.x;
    const int lane  = tid & 31;
    const int wave  = tid >> 5;
    const int waveM = wave >> 1;          // 0..3
    const int waveN = wave & 1;           // 0..1
    const int hi    = lane >> 4;
    const int ml    = lane & 15;
    const int bm    = blockIdx.x * BM;
    const int bn    = blockIdx.y * BN;

    // per-thread staging coordinates
    const int ar = tid >> 2;              // A row (chunks at ar + i*64)
    const int ac = (tid & 3) * 8;         // A 8-bf16 chunk column
    const int wr = tid >> 3;              // W row (chunks at wr + i*32)
    const int wc = (tid & 7) * 4;         // W 4-f32 chunk column

    v8f acc[4][4] = {};

    // ---- prologue: stage K-tile 0 into buffer 0 ----
#pragma unroll
    for (int i = 0; i < 4; ++i)
        async_ld_b128(&As[0][ar + i * 64][ac],
                      &A[(size_t)(bm + ar + i * 64) * GK + ac]);
#pragma unroll
    for (int i = 0; i < 4; ++i) {
        int r = wr + i * 32;
        float4 w = *(const float4*)&W[(size_t)(bn + r) * GK + wc];
        v4bf pk;
        pk[0] = (__bf16)w.x; pk[1] = (__bf16)w.y;
        pk[2] = (__bf16)w.z; pk[3] = (__bf16)w.w;
        *(v4bf*)&Bs[0][r][wc] = pk;
    }
    wait_async0();
    __syncthreads();

    constexpr int NK = GK / BK;
    for (int kt = 0; kt < NK; ++kt) {
        const int cur = kt & 1;
        const int nxt = cur ^ 1;
        const int knext = (kt + 1) * BK;
        float4 wreg[4];

        // ---- prefetch tile kt+1 (async A -> LDS, W -> regs) ----
        if (kt + 1 < NK) {
#pragma unroll
            for (int i = 0; i < 4; ++i)
                async_ld_b128(&As[nxt][ar + i * 64][ac],
                              &A[(size_t)(bm + ar + i * 64) * GK + knext + ac]);
#pragma unroll
            for (int i = 0; i < 4; ++i)
                wreg[i] = *(const float4*)&W[(size_t)(bn + wr + i * 32) * GK + knext + wc];
        }

        // ---- compute on current buffer: 16 WMMAs, 4x fragment reuse ----
        v16bf afrag[4], bfrag[4];
#pragma unroll
        for (int im = 0; im < 4; ++im)
            afrag[im] = load_frag_row(&As[cur][waveM * 64 + im * 16 + ml][0], hi);
#pragma unroll
        for (int in = 0; in < 4; ++in)
            bfrag[in] = load_frag_row(&Bs[cur][waveN * 64 + in * 16 + ml][0], hi);
#pragma unroll
        for (int im = 0; im < 4; ++im)
#pragma unroll
            for (int in = 0; in < 4; ++in)
                acc[im][in] = __builtin_amdgcn_wmma_f32_16x16x32_bf16(
                    false, afrag[im], false, bfrag[in],
                    (short)0, acc[im][in], false, false);

        // ---- drain W regs into next buffer (packed b64), then sync ----
        if (kt + 1 < NK) {
#pragma unroll
            for (int i = 0; i < 4; ++i) {
                v4bf pk;
                pk[0] = (__bf16)wreg[i].x; pk[1] = (__bf16)wreg[i].y;
                pk[2] = (__bf16)wreg[i].z; pk[3] = (__bf16)wreg[i].w;
                *(v4bf*)&Bs[nxt][wr + i * 32][wc] = pk;
            }
        }
        wait_async0();
        __syncthreads();
    }

    // ---- epilogue: constant-stride stores (immediate offsets) ----
#pragma unroll
    for (int im = 0; im < 4; ++im)
#pragma unroll
        for (int in = 0; in < 4; ++in) {
            size_t m0 = (size_t)(bm + waveM * 64 + im * 16 + hi * 8);
            size_t n  = (size_t)(bn + waveN * 64 + in * 16 + ml);
            if (STORE_F32) {
                float* p = (float*)outp + m0 * GN + n;
#pragma unroll
                for (int r = 0; r < 8; ++r) p[(size_t)r * GN] = acc[im][in][r];
            } else {
                __bf16* p = (__bf16*)outp + m0 * GN + n;
#pragma unroll
                for (int r = 0; r < 8; ++r) p[(size_t)r * GN] = (__bf16)acc[im][in][r];
            }
        }
}

// ---------------------------------------------------------------------------
// RoPE applied in-place to Q and K (bf16, layout [token, h*128+d]).
// pair (d, d+64), theta = pos * 10000^(-d/64)
// ---------------------------------------------------------------------------
__global__ __launch_bounds__(256)
void rope_inplace(__bf16* __restrict__ q, __bf16* __restrict__ k) {
    int idx = blockIdx.x * 256 + threadIdx.x;   // QLEN*NH*64 threads
    int d = idx & 63;
    int h = (idx >> 6) & 31;
    int t = idx >> 11;
    float inv = __expf((float)d * -0.14391156847f);   // 10000^(-d/64)
    float ang = (float)t * inv;
    float s, c;
    __sincosf(ang, &s, &c);
    size_t base = (size_t)t * HID + h * DH + d;
    float x1 = (float)q[base], x2 = (float)q[base + 64];
    q[base]      = (__bf16)(x1 * c - x2 * s);
    q[base + 64] = (__bf16)(x2 * c + x1 * s);
    x1 = (float)k[base]; x2 = (float)k[base + 64];
    k[base]      = (__bf16)(x1 * c - x2 * s);
    k[base + 64] = (__bf16)(x2 * c + x1 * s);
}

// ---------------------------------------------------------------------------
// Flash attention, causal.  Block = (q-tile of 64 rows, head), 4 waves.
// Each wave owns 16 q-rows: S(16x64) via QK^T, online softmax within wave,
// P staged through per-wave LDS patch, O(16x128) via P@V.
// K tile staged with async-to-LDS; V staged transposed through registers.
// ---------------------------------------------------------------------------
__global__ __launch_bounds__(128)
void flash_attn(const __bf16* __restrict__ q, const __bf16* __restrict__ kk,
                const __bf16* __restrict__ vv, __bf16* __restrict__ o) {
    __shared__ __bf16 Kt[64][DH + 8];        // keys row-major   (stride 136 = 272B)
    __shared__ __bf16 Vt[DH][64 + 8];        // V transposed     (stride 72  = 144B)
    __shared__ __bf16 Pt[4][16][64 + 8];     // per-wave P patch

    const int tid  = threadIdx.x;
    const int lane = tid & 31;
    const int wave = tid >> 5;
    const int hi   = lane >> 4;
    const int ln   = lane & 15;
    const int qt   = blockIdx.x;             // 0..31
    const int h    = blockIdx.y;             // 0..31
    const int rowbase = qt * 64 + wave * 16;

    // Q operand fragments for all 4 K-steps of depth 32 (d = 0..127)
    v16bf aq[4];
#pragma unroll
    for (int ks = 0; ks < 4; ++ks) {
        const __bf16* p = q + (size_t)(rowbase + ln) * HID + h * DH + ks * 32;
        union { v16bf v; v8bf hh[2]; } u;
        u.hh[0] = *(const v8bf*)(p + hi * 8);
        u.hh[1] = *(const v8bf*)(p + 16 + hi * 8);
        aq[ks] = u.v;
    }

    v8f oacc[8] = {};
    float mrow[8], lrow[8];
#pragma unroll
    for (int r = 0; r < 8; ++r) { mrow[r] = -3.0e38f; lrow[r] = 0.0f; }

    for (int kt = 0; kt <= qt; ++kt) {
        // --- stage K tile (async) and transposed V tile (regs) ---
#pragma unroll
        for (int i = 0; i < 8; ++i) {
            int c  = tid + i * 128;
            int r  = c >> 4;
            int cg = (c & 15) * 8;
            async_ld_b128(&Kt[r][cg],
                          &kk[(size_t)(kt * 64 + r) * HID + h * DH + cg]);
        }
#pragma unroll
        for (int i = 0; i < 8; ++i) {
            int c  = tid + i * 128;
            int r  = c >> 4;
            int cg = (c & 15) * 8;
            v8bf vload = *(const v8bf*)&vv[(size_t)(kt * 64 + r) * HID + h * DH + cg];
#pragma unroll
            for (int j = 0; j < 8; ++j) Vt[cg + j][r] = vload[j];
        }
        wait_async0();
        __syncthreads();

        // --- S = Q K^T  (16 rows x 64 keys per wave) ---
        v8f sacc[4] = {};
#pragma unroll
        for (int in = 0; in < 4; ++in)
#pragma unroll
            for (int ks = 0; ks < 4; ++ks) {
                v16bf bf = load_frag_row(&Kt[in * 16 + ln][ks * 32], hi);
                sacc[in] = __builtin_amdgcn_wmma_f32_16x16x32_bf16(
                    false, aq[ks], false, bf, (short)0, sacc[in], false, false);
            }

        // --- scale + causal mask + tile row max ---
        const float scale = 0.08838834764831845f;   // 1/sqrt(128)
        float tmax[8];
#pragma unroll
        for (int r = 0; r < 8; ++r) tmax[r] = -3.0e38f;
#pragma unroll
        for (int in = 0; in < 4; ++in)
#pragma unroll
            for (int r = 0; r < 8; ++r) {
                float sv  = sacc[in][r] * scale;
                int qrow  = rowbase + hi * 8 + r;
                int key   = kt * 64 + in * 16 + ln;
                if (key > qrow) sv = -1.0e9f;
                sacc[in][r] = sv;
                tmax[r] = fmaxf(tmax[r], sv);
            }
#pragma unroll
        for (int msk = 1; msk < 16; msk <<= 1)
#pragma unroll
            for (int r = 0; r < 8; ++r)
                tmax[r] = fmaxf(tmax[r], __shfl_xor(tmax[r], msk, 32));

        float alpha[8];
#pragma unroll
        for (int r = 0; r < 8; ++r) {
            float mn  = fmaxf(mrow[r], tmax[r]);
            alpha[r]  = __expf(mrow[r] - mn);
            mrow[r]   = mn;
        }

        // --- P = exp(S - m), row sums, stash P to per-wave LDS patch ---
        float rsum[8];
#pragma unroll
        for (int r = 0; r < 8; ++r) rsum[r] = 0.0f;
#pragma unroll
        for (int in = 0; in < 4; ++in)
#pragma unroll
            for (int r = 0; r < 8; ++r) {
                float pv = __expf(sacc[in][r] - mrow[r]);
                rsum[r] += pv;
                Pt[wave][hi * 8 + r][in * 16 + ln] = (__bf16)pv;
            }
#pragma unroll
        for (int msk = 1; msk < 16; msk <<= 1)
#pragma unroll
            for (int r = 0; r < 8; ++r)
                rsum[r] += __shfl_xor(rsum[r], msk, 32);
#pragma unroll
        for (int r = 0; r < 8; ++r) lrow[r] = lrow[r] * alpha[r] + rsum[r];

        // --- rescale O, then O += P @ V ---
#pragma unroll
        for (int jn = 0; jn < 8; ++jn)
#pragma unroll
            for (int r = 0; r < 8; ++r) oacc[jn][r] *= alpha[r];

#pragma unroll
        for (int ks = 0; ks < 2; ++ks) {          // keys depth 64 = 2 x 32
            v16bf pf = load_frag_row(&Pt[wave][ln][ks * 32], hi);
#pragma unroll
            for (int jn = 0; jn < 8; ++jn) {      // output d = 8 x 16
                v16bf vf = load_frag_row(&Vt[jn * 16 + ln][ks * 32], hi);
                oacc[jn] = __builtin_amdgcn_wmma_f32_16x16x32_bf16(
                    false, pf, false, vf, (short)0, oacc[jn], false, false);
            }
        }
        __syncthreads();   // before overwriting Kt/Vt next iteration
    }

    // --- O /= l, write bf16 attn output [token, h*128 + d] ---
#pragma unroll
    for (int jn = 0; jn < 8; ++jn) {
        __bf16* p = o + (size_t)(rowbase + hi * 8) * HID + h * DH + jn * 16 + ln;
#pragma unroll
        for (int r = 0; r < 8; ++r)
            p[(size_t)r * HID] = (__bf16)(oacc[jn][r] / lrow[r]);
    }
}

// ---------------------------------------------------------------------------
// Launch: cvt -> 3x GEMM (QKV) -> RoPE -> flash attention -> out projection
// ---------------------------------------------------------------------------
extern "C" void kernel_launch(void* const* d_in, const int* in_sizes, int n_in,
                              void* d_out, int out_size, void* d_ws, size_t ws_size,
                              hipStream_t stream) {
    const float* hidden = (const float*)d_in[0];
    // d_in[1] = position_ids (arange, implicit), d_in[2] = causal mask (implicit)
    const float* wq = (const float*)d_in[3];
    const float* wk = (const float*)d_in[4];
    const float* wv = (const float*)d_in[5];
    const float* wo = (const float*)d_in[6];

    const size_t SZ = (size_t)QLEN * HID;           // 8,388,608 elements
    __bf16* hb = (__bf16*)d_ws;                     // hidden   bf16
    __bf16* qb = hb + SZ;                           // Q        bf16
    __bf16* kb = qb + SZ;                           // K        bf16
    __bf16* vb = kb + SZ;                           // V        bf16
    __bf16* ab = vb + SZ;                           // attn out bf16

    cvt_f32_to_bf16<<<(unsigned)(SZ / (256 * 8)), 256, 0, stream>>>(hidden, hb);

    dim3 gg(QLEN / BM, HID / BN);                   // (8, 32)
    gemm_bf16<false><<<gg, 256, 0, stream>>>(hb, wq, qb);
    gemm_bf16<false><<<gg, 256, 0, stream>>>(hb, wk, kb);
    gemm_bf16<false><<<gg, 256, 0, stream>>>(hb, wv, vb);

    rope_inplace<<<(QLEN * NH * 64) / 256, 256, 0, stream>>>(qb, kb);

    dim3 ga(QLEN / 64, NH);                         // (32, 32)
    flash_attn<<<ga, 128, 0, stream>>>(qb, kb, vb, ab);

    gemm_bf16<true><<<gg, 256, 0, stream>>>(ab, wo, (float*)d_out);
}
